// NSLoss_386547057230
// MI455X (gfx1250) — matrix operand order repaired and checked by reading, not compiled, not run on hardware
//
#include <hip/hip_runtime.h>

// CDNA5 / gfx1250 wave32 WMMA types
typedef __attribute__((ext_vector_type(16))) _Float16 v16h;
typedef __attribute__((ext_vector_type(8)))  _Float16 v8h;
typedef __attribute__((ext_vector_type(4)))  _Float16 v4h;
typedef __attribute__((ext_vector_type(8)))  float    v8f;

#define WPB      4      // waves per block (128 threads)
#define TILE_M   16     // batch rows per wave tile
#define DIM      128    // embedding size
#define RSTRIDE  136    // LDS row stride in halves (272 B) -> conflict-free banks

union V16u { v16h v; v8h half8[2]; };

__global__ void nsloss_zero_kernel(float* out) { out[0] = 0.0f; }

// stable log_sigmoid: min(x,0) - log(1 + exp(-|x|)); arg of log is in (1,2],
// so fast v_log_f32 is numerically fine (no log1p-critical region).
__device__ __forceinline__ float log_sigmoid_fast(float x) {
    return fminf(x, 0.0f) - __logf(1.0f + __expf(-fabsf(x)));
}

__global__ __launch_bounds__(WPB * 32)
void nsloss_wmma_kernel(const float* __restrict__ emb,
                        const float* __restrict__ weights,
                        const long long* __restrict__ label,
                        const long long* __restrict__ negs,
                        float* __restrict__ out,
                        int n, int num_sampled, float neg_inv_n)
{
    __shared__ __align__(16) _Float16 lds_e[WPB][TILE_M * RSTRIDE];
    __shared__ __align__(16) _Float16 lds_w[WPB][TILE_M * RSTRIDE];

    const int lane = threadIdx.x & 31;
    const int wsub = threadIdx.x >> 5;
    const int ntiles = n >> 4;

    int tile = blockIdx.x * WPB + wsub;
    tile = __builtin_amdgcn_readfirstlane(tile);   // wave-uniform -> scalar path
    if (tile >= ntiles) return;                    // whole-wave exit (EXEC stays all-1s)

    const long long i0 = (long long)tile << 4;
    _Float16* eL = lds_e[wsub];
    _Float16* wL = lds_w[wsub];

    // ---- Stage 16 emb rows into LDS as f16 (coalesced b128 gathers, cvt in flight) ----
    #pragma unroll 4
    for (int r = 0; r < TILE_M; ++r) {
        const float4* er = reinterpret_cast<const float4*>(emb + (i0 + r) * DIM);
        float4 v = er[lane];
        v4h h; h.x = (_Float16)v.x; h.y = (_Float16)v.y;
               h.z = (_Float16)v.z; h.w = (_Float16)v.w;
        *reinterpret_cast<v4h*>(&eL[r * RSTRIDE + lane * 4]) = h;
    }

    // ---- Build the 4 B fragments once (reused across all 11 targets) ----
    // B[k][n] = emb[i0+n][k]; lane n<16 holds K k0..k0+7,k0+16..k0+23, lane n+16 holds K k0+8..k0+15,k0+24..k0+31
    const int bsel = (lane < 16) ? 0 : 8;
    const int bn   = lane & 15;
    V16u bfrag[4];
    #pragma unroll
    for (int s = 0; s < 4; ++s) {
        const _Float16* p = &eL[bn * RSTRIDE + 32 * s + bsel];
        bfrag[s].half8[0] = *reinterpret_cast<const v8h*>(p);
        bfrag[s].half8[1] = *reinterpret_cast<const v8h*>(p + 16);
    }

    const int aoff = (lane < 16) ? 0 : 8;     // A layout half-wave K offset
    const int am   = lane & 15;
    const int ntargets = num_sampled + 1;
    float csum = 0.0f;

    for (int t = 0; t < ntargets; ++t) {
        // ---- Gather 16 weight rows for this target (uniform index -> s_load,
        //      row gather: 32 lanes x b128 = one full 512B row) ----
        #pragma unroll 4
        for (int r = 0; r < TILE_M; ++r) {
            long long row = (t == 0) ? label[i0 + r]
                                     : negs[(i0 + r) * num_sampled + (t - 1)];
            const float4* wr = reinterpret_cast<const float4*>(weights + row * DIM);
            float4 v = wr[lane];
            v4h h; h.x = (_Float16)v.x; h.y = (_Float16)v.y;
                   h.z = (_Float16)v.z; h.w = (_Float16)v.w;
            *reinterpret_cast<v4h*>(&wL[r * RSTRIDE + lane * 4]) = h;
        }

        // ---- 16x16x128 GEMM via 4 chained v_wmma_f32_16x16x32_f16 ----
        v8f acc = {};
        #pragma unroll
        for (int s = 0; s < 4; ++s) {
            V16u af;
            const _Float16* p = &wL[am * RSTRIDE + 32 * s + aoff];
            af.half8[0] = *reinterpret_cast<const v8h*>(p);       // K k0+off .. +7
            af.half8[1] = *reinterpret_cast<const v8h*>(p + 16);  // K k0+16+off .. +7
            acc = __builtin_amdgcn_wmma_f32_16x16x32_f16(
                false, af.v, false, bfrag[s].v, (short)0, acc, false, false);
        }

        // ---- Extract diagonal D[m][m]: (vgpr r, lane r) for m<8; (vgpr r, lane r+24) for m=r+8 ----
        float d = 0.0f;
        #pragma unroll
        for (int r8 = 0; r8 < 8; ++r8) {
            float v = acc[r8];
            bool take = (lane == r8) || (lane == r8 + 24);
            d = take ? v : d;
        }
        const bool active = (lane < 8) || (lane >= 24);
        float score = (t == 0) ? d : -d;                 // negatives use -W
        csum += active ? log_sigmoid_fast(score) : 0.0f;
    }

    // ---- wave32 reduction, one atomic per wave ----
    #pragma unroll
    for (int o = 16; o > 0; o >>= 1) csum += __shfl_xor(csum, o, 32);
    if (lane == 0) atomicAdd(out, csum * neg_inv_n);
}

extern "C" void kernel_launch(void* const* d_in, const int* in_sizes, int n_in,
                              void* d_out, int out_size, void* d_ws, size_t ws_size,
                              hipStream_t stream) {
    // inputs: y_hat[n], emb[n*128], weights[1e6*128], label[n] i64, negs[n*10] i64
    const float*     emb     = (const float*)d_in[1];
    const float*     weights = (const float*)d_in[2];
    const long long* label   = (const long long*)d_in[3];
    const long long* negs    = (const long long*)d_in[4];
    float* out = (float*)d_out;

    const int n  = in_sizes[0];
    const int ns = in_sizes[4] / n;     // num_sampled (10)

    nsloss_zero_kernel<<<1, 1, 0, stream>>>(out);

    const int ntiles = n / 16;                          // N=16384 -> 1024 tiles
    const int blocks = (ntiles + WPB - 1) / WPB;        // 256 blocks x 4 waves
    nsloss_wmma_kernel<<<blocks, WPB * 32, 0, stream>>>(emb, weights, label, negs,
                                                        out, n, ns, -1.0f / (float)n);
}
